// VectorQuantizer_29892972380630
// MI455X (gfx1250) — matrix operand order repaired and checked by reading, not compiled, not run on hardware
//
#include <hip/hip_runtime.h>
#include <hip/hip_bf16.h>

typedef __attribute__((ext_vector_type(16))) __bf16 v16bf;
typedef __attribute__((ext_vector_type(8)))  float  v8f;

#define B_ROWS    65536
#define DIM       256
#define KCODES    1024
#define TILE_ROWS 128          // 8 waves x 16 rows per workgroup
#define WAVES_PER_WG 8

// ---------------------------------------------------------------------------
// Prep: split codebook fp32 -> bf16 (hi, lo), compute ||e||^2, zero loss slot.
// One block per code (k), 256 threads = one per dim.
// ---------------------------------------------------------------------------
__global__ void __launch_bounds__(256) vq_prep(
    const float* __restrict__ codebook,
    __bf16* __restrict__ e_hi, __bf16* __restrict__ e_lo,
    float* __restrict__ e_sq, float* __restrict__ loss_slot) {
  const int k = blockIdx.x;
  const int t = threadIdx.x;
  const float c = codebook[(size_t)k * DIM + t];
  const __bf16 h = (__bf16)c;
  const __bf16 l = (__bf16)(c - (float)h);
  e_hi[(size_t)k * DIM + t] = h;
  e_lo[(size_t)k * DIM + t] = l;

  float s = c * c;
  #pragma unroll
  for (int off = 16; off > 0; off >>= 1) s += __shfl_xor(s, off, 32);

  __shared__ float red[WAVES_PER_WG];
  const int wave = t >> 5, lane = t & 31;
  if (lane == 0) red[wave] = s;
  __syncthreads();
  if (t == 0) {
    float tot = 0.f;
    #pragma unroll
    for (int i = 0; i < WAVES_PER_WG; ++i) tot += red[i];
    e_sq[k] = tot;
    if (k == 0) loss_slot[0] = 0.0f;   // re-init every call (harness poisons once)
  }
}

// ---------------------------------------------------------------------------
// Main: per-workgroup 128 rows. Each wave does 16 rows x all 1024 codes via
// V_WMMA_F32_16X16X32_BF16 in bf16x3 (hi*hi + hi*lo + lo*hi) for ~fp32 dots.
// Then per-row argmin (wave32 shfl reduce), gather + exact fp32 MSE.
// ---------------------------------------------------------------------------
__global__ void __launch_bounds__(256) vq_main(
    const float* __restrict__ x, const float* __restrict__ codebook,
    const __bf16* __restrict__ e_hi, const __bf16* __restrict__ e_lo,
    const float* __restrict__ e_sq,
    float* __restrict__ out_q, float* __restrict__ out_idx,
    float* __restrict__ out_loss) {
  const int tid  = threadIdx.x;
  const int wave = tid >> 5;
  const int lane = tid & 31;
  const int half = lane >> 4;     // wave32: lanes 0-15 / 16-31
  const int lm   = lane & 15;
  const int rowbase = blockIdx.x * TILE_ROWS;
  const int wrow0   = rowbase + wave * 16;

  // ---- Load this wave's 16-row A tile, split to bf16 hi/lo fragments ----
  // A 16x32 layout (matches B): lane holds row lm; element s -> K = 16*half + s
  // (one run of 16 contiguous floats).
  const float* xrow = x + (size_t)(wrow0 + lm) * DIM;
  v16bf a_hi[8], a_lo[8];
  float part = 0.f;
  #pragma unroll
  for (int c = 0; c < 8; ++c) {
    const int base0 = c * 32 + 16 * half;
    const float4 f0 = *(const float4*)(xrow + base0);
    const float4 f1 = *(const float4*)(xrow + base0 + 4);
    const float4 f2 = *(const float4*)(xrow + base0 + 8);
    const float4 f3 = *(const float4*)(xrow + base0 + 12);
    float v[16];
    v[0]=f0.x; v[1]=f0.y; v[2]=f0.z; v[3]=f0.w;
    v[4]=f1.x; v[5]=f1.y; v[6]=f1.z; v[7]=f1.w;
    v[8]=f2.x; v[9]=f2.y; v[10]=f2.z; v[11]=f2.w;
    v[12]=f3.x; v[13]=f3.y; v[14]=f3.z; v[15]=f3.w;
    #pragma unroll
    for (int t = 0; t < 16; ++t) {
      const float f = v[t];
      part += f * f;
      const __bf16 h = (__bf16)f;
      a_hi[c][t] = h;
      a_lo[c][t] = (__bf16)(f - (float)h);
    }
  }

  // ||x||^2 per row: row lm is split across lanes lm and lm+16.
  const float xsq_full = part + __shfl_xor(part, 16, 32);
  // C-tile rows owned by this lane are r = i + 8*half; fetch their ||x||^2.
  float xsq_i[8];
  #pragma unroll
  for (int i = 0; i < 8; ++i)
    xsq_i[i] = __shfl(xsq_full, i + 8 * half, 32);

  float best[8];
  int   bidx[8];
  #pragma unroll
  for (int i = 0; i < 8; ++i) { best[i] = 3.4e38f; bidx[i] = 0; }

  // ---- Sweep all 64 column tiles of 16 codes ----
  for (int ct = 0; ct < KCODES / 16; ++ct) {
    const int col = ct * 16 + lm;   // this lane's code (B-matrix column)
    // B 32x16 layout: lane holds column lm; half selects K 0-15 / 16-31:
    // 16 contiguous bf16 per fragment -> one 32B vector load.
    const __bf16* bh = e_hi + (size_t)col * DIM + 16 * half;
    const __bf16* bl = e_lo + (size_t)col * DIM + 16 * half;

    v8f acc = {};
    #pragma unroll
    for (int c = 0; c < 8; ++c) {
      const v16bf bhv = *(const v16bf*)(bh + c * 32);
      const v16bf blv = *(const v16bf*)(bl + c * 32);
      acc = __builtin_amdgcn_wmma_f32_16x16x32_bf16(false, a_hi[c], false, bhv,
                                                    (short)0, acc, false, false);
      acc = __builtin_amdgcn_wmma_f32_16x16x32_bf16(false, a_hi[c], false, blv,
                                                    (short)0, acc, false, false);
      acc = __builtin_amdgcn_wmma_f32_16x16x32_bf16(false, a_lo[c], false, bhv,
                                                    (short)0, acc, false, false);
    }

    const float esq = e_sq[col];
    #pragma unroll
    for (int i = 0; i < 8; ++i) {
      const float d = xsq_i[i] + esq - 2.0f * acc[i];
      if (d < best[i]) { best[i] = d; bidx[i] = col; }
    }
  }

  // ---- argmin reduce across the 16 lanes of each half (columns vary) ----
  #pragma unroll
  for (int i = 0; i < 8; ++i) {
    float bv = best[i];
    int   bi = bidx[i];
    #pragma unroll
    for (int off = 1; off < 16; off <<= 1) {
      const float ov = __shfl_xor(bv, off, 32);
      const int   oi = __shfl_xor(bi, off, 32);
      if (ov < bv || (ov == bv && oi < bi)) { bv = ov; bi = oi; }
    }
    bidx[i] = bi;
  }

  __shared__ int s_idx[TILE_ROWS];
  if (lm == 0) {   // lanes 0 and 16 of each wave publish their 8 rows
    #pragma unroll
    for (int i = 0; i < 8; ++i) {
      const int r = wave * 16 + 8 * half + i;
      s_idx[r] = bidx[i];
      out_idx[rowbase + r] = (float)bidx[i];
    }
  }
  __syncthreads();

  // ---- Phase 2: gather codebook[idx] (coalesced: thread = dim), write
  // quantized (STE output == codebook row), exact fp32 commitment MSE. ----
  float lsum = 0.f;
  const int d = tid;   // 0..255 == dim
  for (int r = 0; r < TILE_ROWS; ++r) {
    const int idx = s_idx[r];
    const float q  = codebook[(size_t)idx * DIM + d];
    const size_t off = (size_t)(rowbase + r) * DIM + d;
    const float xv = x[off];
    out_q[off] = q;
    const float df = xv - q;
    lsum += df * df;
  }
  #pragma unroll
  for (int off = 16; off > 0; off >>= 1) lsum += __shfl_xor(lsum, off, 32);
  __shared__ float s_red[WAVES_PER_WG];
  if (lane == 0) s_red[wave] = lsum;
  __syncthreads();
  if (tid == 0) {
    float tot = 0.f;
    #pragma unroll
    for (int i = 0; i < WAVES_PER_WG; ++i) tot += s_red[i];
    atomicAdd(out_loss, tot * (1.0f / ((float)B_ROWS * (float)DIM)));
  }
}

// ---------------------------------------------------------------------------
extern "C" void kernel_launch(void* const* d_in, const int* in_sizes, int n_in,
                              void* d_out, int out_size, void* d_ws, size_t ws_size,
                              hipStream_t stream) {
  const float* x        = (const float*)d_in[0];
  const float* codebook = (const float*)d_in[1];

  float* out      = (float*)d_out;
  float* out_q    = out;                                 // [B, D]
  float* out_idx  = out + (size_t)B_ROWS * DIM;          // [B] (as float)
  float* out_loss = out_idx + B_ROWS;                    // scalar

  // workspace: e_hi (512KB) | e_lo (512KB) | e_sq (4KB)  ~= 1.05 MB
  __bf16* e_hi = (__bf16*)d_ws;
  __bf16* e_lo = e_hi + (size_t)KCODES * DIM;
  float*  e_sq = (float*)(e_lo + (size_t)KCODES * DIM);

  vq_prep<<<KCODES, 256, 0, stream>>>(codebook, e_hi, e_lo, e_sq, out_loss);
  vq_main<<<B_ROWS / TILE_ROWS, 256, 0, stream>>>(
      x, codebook, e_hi, e_lo, e_sq, out_q, out_idx, out_loss);
}
